// DC_GCN_67250597921035
// MI455X (gfx1250) — matrix operand order repaired and checked by reading, not compiled, not run on hardware
//
#include <hip/hip_runtime.h>
#include <hip/hip_bf16.h>

// ---------------------------------------------------------------------------
// Types for CDNA5 WMMA (wave32): bf16 A/B fragments (16 x bf16 = 8 VGPRs),
// f32 accumulator (8 x f32 = 8 VGPRs).
// ---------------------------------------------------------------------------
typedef __attribute__((ext_vector_type(16))) __bf16        v16bf;
typedef __attribute__((ext_vector_type(8)))  unsigned int  v8ui;
typedef __attribute__((ext_vector_type(8)))  float         v8f;

__device__ __forceinline__ unsigned int f2bf_bits(float f) {
    // round-to-nearest-even fp32 -> bf16 (result in low 16 bits)
    unsigned int u = __float_as_uint(f);
    unsigned int r = u + 0x7fffu + ((u >> 16) & 1u);
    return r >> 16;
}

__device__ __forceinline__ float sigmoidf(float x) {
    return 1.0f / (1.0f + __expf(-x));
}

// ---------------------------------------------------------------------------
// WMMA GEMM: C[M,N] = act(A[M,K] @ B[K,N] + bias), fp32 in/out, bf16 compute,
// fp32 accumulate via v_wmma_f32_16x16x32_bf16.
// One 16x16 C tile per wave; 4 waves per 128-thread block.
// Requires K % 32 == 0, M % 64 == 0. N handled branchlessly (clamp + mask).
// ---------------------------------------------------------------------------
__global__ void gemm_wmma_bf16(const float* __restrict__ A,
                               const float* __restrict__ B,
                               const float* __restrict__ bias,
                               float* __restrict__ C,
                               int M, int K, int N, int do_relu)
{
    const int lane = threadIdx.x & 31;
    const int wave = threadIdx.x >> 5;
    const int tm   = blockIdx.y * 4 + wave;   // 16-row tile index (wave-uniform)
    if (tm * 16 >= M) return;                 // uniform per wave: EXEC stays all-1s
    const int tn   = blockIdx.x;              // 16-col tile index
    const int row0 = tm * 16;
    const int half = lane >> 4;               // K half-split selector
    const int r    = lane & 15;
    const int colB = tn * 16 + r;             // B-fragment / C-store column

    // Branchless OOB handling for the N=40 tail tile: load at a clamped
    // in-bounds column, zero the converted bits with a mask.
    const int          cs   = (colB < N) ? colB : 0;
    const unsigned int keep = (colB < N) ? 0xffffffffu : 0u;

    const float* __restrict__ Arow = A + (size_t)(row0 + r) * K;   // this lane's A row
    const float* __restrict__ Bcol = B + cs;                       // this lane's B column

    v8f acc = {};
    for (int kk = 0; kk < K; kk += 32) {
        // Prefetch next K-slab of this lane's A row (global_prefetch_b8).
        if (kk + 32 < K) __builtin_prefetch(Arow + kk + 32, 0, 1);

        v8ui ua, ub;
        #pragma unroll
        for (int v = 0; v < 8; ++v) {
            // A 16x32 bf16 layout: lanes 0-15 rows M=0..15 with K=0..15,
            // lanes 16-31 same rows with K=16..31 (matches B).
            // K-pairs (k, k+1) are contiguous in row-major A -> float2 load.
            const int ka = kk + (v << 1) + (half << 4);
            const float2 af = *(const float2*)(Arow + ka);
            ua[v] = f2bf_bits(af.x) | (f2bf_bits(af.y) << 16);

            // B 32x16 bf16 layout: lane = column; lanes 0-15 hold K=0..15,
            // lanes 16-31 hold K=16..31; each VGPR packs a K pair.
            const int kb = kk + (v << 1) + (half << 4);
            const float b0 = Bcol[(size_t)kb * N];
            const float b1 = Bcol[(size_t)(kb + 1) * N];
            ub[v] = (f2bf_bits(b0) | (f2bf_bits(b1) << 16)) & keep;
        }
        v16bf a = __builtin_bit_cast(v16bf, ua);
        v16bf b = __builtin_bit_cast(v16bf, ub);
        acc = __builtin_amdgcn_wmma_f32_16x16x32_bf16(
                  false, a, false, b, (short)0, acc, false, false);
    }

    if (colB < N) {
        const float bv = bias ? bias[colB] : 0.0f;
        #pragma unroll
        for (int v = 0; v < 8; ++v) {
            const int row = row0 + v + (half << 3);   // C/D layout: VGPR v -> M
            float val = acc[v] + bv;
            if (do_relu) val = fmaxf(val, 0.0f);
            C[(size_t)row * N + colB] = val;
        }
    }
}

// ---------------------------------------------------------------------------
// MLP head: out[i] = sigmoid(dot(H[i,:K], w2) + b2).  One wave per row.
// ---------------------------------------------------------------------------
__global__ void mlp_head(const float* __restrict__ H,
                         const float* __restrict__ w2,
                         const float* __restrict__ b2,
                         float* __restrict__ out, int K)
{
    const int lane = threadIdx.x & 31;
    const int wid  = threadIdx.x >> 5;
    const int row  = blockIdx.x * (blockDim.x >> 5) + wid;
    float acc = 0.0f;
    for (int k = lane; k < K; k += 32)
        acc += H[(size_t)row * K + k] * w2[k];
    for (int off = 16; off; off >>= 1)
        acc += __shfl_xor(acc, off, 32);
    if (lane == 0)
        out[row] = sigmoidf(acc + b2[0]);
}

// ---------------------------------------------------------------------------
// Curvature pass 1: per row i, fused over w = adj * sigmoid(weights):
//   gamma_f[i] = 0.5 * sum_j w*fd^2,  delta_f[i] = sum_j w*fd,  fd = f[j]-f[i]
// ---------------------------------------------------------------------------
__global__ void curv_pass1(const float* __restrict__ adj,
                           const float* __restrict__ wts,
                           const float* __restrict__ f,
                           float* __restrict__ gamma_f,
                           float* __restrict__ delta_f, int n)
{
    const int i = blockIdx.x;
    const float fi = f[i];
    float g = 0.0f, d = 0.0f;
    const size_t base = (size_t)i * n;
    for (int j = threadIdx.x; j < n; j += blockDim.x) {
        const float a = adj[base + j];
        if (a != 0.0f) {
            const float w  = a * sigmoidf(wts[base + j]);
            const float fd = f[j] - fi;
            g += w * fd * fd;
            d += w * fd;
        }
    }
    __shared__ float sg[8], sd[8];
    for (int off = 16; off; off >>= 1) {
        g += __shfl_xor(g, off, 32);
        d += __shfl_xor(d, off, 32);
    }
    const int wid = threadIdx.x >> 5, lane = threadIdx.x & 31;
    if (lane == 0) { sg[wid] = g; sd[wid] = d; }
    __syncthreads();
    if (threadIdx.x == 0) {
        float G = 0.0f, D = 0.0f;
        const int nw = blockDim.x >> 5;
        for (int t = 0; t < nw; ++t) { G += sg[t]; D += sd[t]; }
        gamma_f[i] = 0.5f * G;
        delta_f[i] = D;
    }
}

// ---------------------------------------------------------------------------
// Curvature pass 2: gamma2[i] = 0.5*delta_gamma[i] - 0.5*sum_j w*fd*dfd
// ---------------------------------------------------------------------------
__global__ void curv_pass2(const float* __restrict__ adj,
                           const float* __restrict__ wts,
                           const float* __restrict__ f,
                           const float* __restrict__ gamma_f,
                           const float* __restrict__ delta_f,
                           float* __restrict__ gamma2, int n)
{
    const int i = blockIdx.x;
    const float fi = f[i], gfi = gamma_f[i], dfi = delta_f[i];
    float dg = 0.0f, gf = 0.0f;
    const size_t base = (size_t)i * n;
    for (int j = threadIdx.x; j < n; j += blockDim.x) {
        const float a = adj[base + j];
        if (a != 0.0f) {
            const float w  = a * sigmoidf(wts[base + j]);
            const float fd = f[j] - fi;
            dg += w * (gamma_f[j] - gfi);
            gf += w * fd * (delta_f[j] - dfi);
        }
    }
    __shared__ float sg[8], sd[8];
    for (int off = 16; off; off >>= 1) {
        dg += __shfl_xor(dg, off, 32);
        gf += __shfl_xor(gf, off, 32);
    }
    const int wid = threadIdx.x >> 5, lane = threadIdx.x & 31;
    if (lane == 0) { sg[wid] = dg; sd[wid] = gf; }
    __syncthreads();
    if (threadIdx.x == 0) {
        float DG = 0.0f, GF = 0.0f;
        const int nw = blockDim.x >> 5;
        for (int t = 0; t < nw; ++t) { DG += sg[t]; GF += sd[t]; }
        gamma2[i] = 0.5f * DG - 0.5f * GF;
    }
}

// ---------------------------------------------------------------------------
// curv_loss partials: part[j] = sum_i relu(kappa[i]*gamma_f[j] - gamma2[j])
// ---------------------------------------------------------------------------
__global__ void curv_loss_partial(const float* __restrict__ kappa,
                                  const float* __restrict__ gamma_f,
                                  const float* __restrict__ gamma2,
                                  float* __restrict__ part, int n)
{
    const int j = blockIdx.x * blockDim.x + threadIdx.x;
    if (j >= n) return;
    const float gj = gamma_f[j], mj = gamma2[j];
    float acc = 0.0f;
    for (int i = 0; i < n; ++i)
        acc += fmaxf(kappa[i] * gj - mj, 0.0f);
    part[j] = acc;
}

// Deterministic single-block reduction (fixed order, no float atomics).
__global__ void reduce_sum(const float* __restrict__ part,
                           float* __restrict__ out, int n)
{
    __shared__ float s[256];
    float a = 0.0f;
    for (int i = threadIdx.x; i < n; i += 256) a += part[i];
    s[threadIdx.x] = a;
    __syncthreads();
    for (int off = 128; off; off >>= 1) {
        if (threadIdx.x < off) s[threadIdx.x] += s[threadIdx.x + off];
        __syncthreads();
    }
    if (threadIdx.x == 0) out[0] = s[0];
}

// ---------------------------------------------------------------------------
// top-k mask via rank (matches jax top_k: descending, ties by lower index).
// ---------------------------------------------------------------------------
__global__ void fill_ones(float* __restrict__ m, int n)
{
    const int i = blockIdx.x * blockDim.x + threadIdx.x;
    if (i < n) m[i] = 1.0f;
}

__global__ void topk_mask(const float* __restrict__ kappa,
                          float* __restrict__ mask, int n, int num)
{
    const int i = blockIdx.x * blockDim.x + threadIdx.x;
    if (i >= n) return;
    const float ki = kappa[i];
    int cnt = 0;
    for (int j = 0; j < n; ++j) {
        const float kj = kappa[j];
        if (kj > ki || (kj == ki && j < i)) ++cnt;
    }
    mask[i] = (cnt < num) ? 0.0f : 1.0f;   // pruned nodes -> 0
}

// ---------------------------------------------------------------------------
// GCN symmetric norm: dinv[j] = (1 + mask[j]*sum_i adj[i,j]*mask[i])^-0.5
// Column sums are coalesced across consecutive j threads; adj is L2-resident.
// ---------------------------------------------------------------------------
__global__ void degree_kernel(const float* __restrict__ adj,
                              const float* __restrict__ mask,
                              float* __restrict__ dinv, int n)
{
    const int j = blockIdx.x * blockDim.x + threadIdx.x;
    if (j >= n) return;
    float s = 0.0f;
    for (int i = 0; i < n; ++i)
        s += adj[(size_t)i * n + j] * mask[i];
    dinv[j] = rsqrtf(1.0f + mask[j] * s);
}

// ---------------------------------------------------------------------------
// Aggregation: out[i,c] = relu( dinv_i*( sum_j Ahat[j,i]*dinv_j*Z[j,c] ) + b[c] )
// with Ahat[j,i] = adj[j,i]*mask_i*mask_j + delta_ij. One block per row i;
// the adj column scan is block-uniform (scalarizable, branch mostly skipped).
// ---------------------------------------------------------------------------
__global__ void aggregate(const float* __restrict__ adj,
                          const float* __restrict__ mask,
                          const float* __restrict__ dinv,
                          const float* __restrict__ Z,
                          const float* __restrict__ bias,
                          float* __restrict__ out,
                          int n, int C, int do_relu)
{
    const int i  = blockIdx.x;
    const int c  = threadIdx.x;
    const float di = dinv[i];
    const float mi = mask[i];
    float acc = (c < C) ? di * Z[(size_t)i * C + c] : 0.0f;   // self loop
    for (int j = 0; j < n; ++j) {
        const float a = adj[(size_t)j * n + i];               // uniform load
        if (a != 0.0f) {
            const float w = mi * mask[j] * a * dinv[j];
            if (w != 0.0f && c < C)
                acc += w * Z[(size_t)j * C + c];
        }
    }
    if (c < C) {
        float val = di * acc + bias[c];
        if (do_relu) val = fmaxf(val, 0.0f);
        out[(size_t)i * C + c] = val;
    }
}

// ---------------------------------------------------------------------------
// Row-wise log_softmax, C <= blockDim (64 threads for C=40).
// ---------------------------------------------------------------------------
__global__ void log_softmax_k(const float* __restrict__ H,
                              float* __restrict__ out, int C)
{
    const int i = blockIdx.x;
    const int c = threadIdx.x;
    __shared__ float red[64];
    const float v = (c < C) ? H[(size_t)i * C + c] : -3.0e38f;
    red[c] = v;
    __syncthreads();
    for (int off = 32; off; off >>= 1) {
        if (c < off) red[c] = fmaxf(red[c], red[c + off]);
        __syncthreads();
    }
    const float m = red[0];
    __syncthreads();
    red[c] = (c < C) ? __expf(v - m) : 0.0f;
    __syncthreads();
    for (int off = 32; off; off >>= 1) {
        if (c < off) red[c] += red[c + off];
        __syncthreads();
    }
    const float lse = __logf(red[0]);
    if (c < C) out[(size_t)i * C + c] = v - m - lse;
}

// ---------------------------------------------------------------------------
// Orchestration
// ---------------------------------------------------------------------------
extern "C" void kernel_launch(void* const* d_in, const int* in_sizes, int n_in,
                              void* d_out, int out_size, void* d_ws, size_t ws_size,
                              hipStream_t stream)
{
    constexpr int N = 4096, NFEAT = 512, NHID = 256, NCLASS = 40;
    constexpr int NUM_PRUNE = (N * 40 * 1) / 100;   // int(4096*p*t/100), t=1 -> 1638

    const float* x   = (const float*)d_in[0];
    const float* adj = (const float*)d_in[1];
    const float* wts = (const float*)d_in[2];
    const float* cw1 = (const float*)d_in[3];
    const float* cb1 = (const float*)d_in[4];
    const float* cw2 = (const float*)d_in[5];
    const float* cb2 = (const float*)d_in[6];
    const float* fw1 = (const float*)d_in[7];
    const float* fb1 = (const float*)d_in[8];
    const float* fw2 = (const float*)d_in[9];
    const float* fb2 = (const float*)d_in[10];
    const float* W0  = (const float*)d_in[11];
    const float* b0  = (const float*)d_in[12];
    const float* W1  = (const float*)d_in[13];
    const float* b1  = (const float*)d_in[14];
    (void)d_in[15]; (void)in_sizes; (void)n_in; (void)out_size; (void)ws_size;

    float* out = (float*)d_out;                 // [N*NCLASS] log_softmax, then loss
    float* ws  = (float*)d_ws;

    // workspace layout (fp32 elements), ~9.2 MB total
    float* HBUF   = ws;                         // N*NHID   (hidden / h)
    float* ZBUF   = ws + 1048576;               // N*NHID   (x@W)
    float* H2     = ws + 2097152;               // N*NCLASS (pre-softmax)
    float* KAPPA  = ws + 2260992;               // N
    float* FVEC   = ws + 2265088;               // N
    float* GAMMAF = ws + 2269184;               // N
    float* DELTAF = ws + 2273280;               // N
    float* GAMMA2 = ws + 2277376;               // N
    float* MASK   = ws + 2281472;               // N
    float* DINV   = ws + 2285568;               // N
    float* PART   = ws + 2289664;               // N

    const dim3 gemmBlk(128);                                  // 4 waves / block
    const dim3 gemmGridWide(NHID / 16, N / 64);               // N-out = 256
    const dim3 gemmGridCls((NCLASS + 15) / 16, N / 64);       // N-out = 40

    // --- kappa = sigmoid MLP(x) ---
    gemm_wmma_bf16<<<gemmGridWide, gemmBlk, 0, stream>>>(x, cw1, cb1, HBUF,
                                                         N, NFEAT, NHID, 1);
    mlp_head<<<N / 8, 256, 0, stream>>>(HBUF, cw2, cb2, KAPPA, NHID);

    // --- f = sigmoid MLP(x) ---
    gemm_wmma_bf16<<<gemmGridWide, gemmBlk, 0, stream>>>(x, fw1, fb1, HBUF,
                                                         N, NFEAT, NHID, 1);
    mlp_head<<<N / 8, 256, 0, stream>>>(HBUF, fw2, fb2, FVEC, NHID);

    // --- curvature terms: two fused passes over adj/weights ---
    curv_pass1<<<N, 256, 0, stream>>>(adj, wts, FVEC, GAMMAF, DELTAF, N);
    curv_pass2<<<N, 256, 0, stream>>>(adj, wts, FVEC, GAMMAF, DELTAF, GAMMA2, N);

    // --- curv_loss (deterministic two-stage reduction) ---
    curv_loss_partial<<<N / 256, 256, 0, stream>>>(KAPPA, GAMMAF, GAMMA2, PART, N);
    reduce_sum<<<1, 256, 0, stream>>>(PART, out + (size_t)N * NCLASS, N);

    // --- GCN layer 0: no pruning ---
    fill_ones<<<N / 256, 256, 0, stream>>>(MASK, N);
    degree_kernel<<<N / 256, 256, 0, stream>>>(adj, MASK, DINV, N);
    gemm_wmma_bf16<<<gemmGridWide, gemmBlk, 0, stream>>>(x, W0, nullptr, ZBUF,
                                                         N, NFEAT, NHID, 0);
    aggregate<<<N, 256, 0, stream>>>(adj, MASK, DINV, ZBUF, b0, HBUF, N, NHID, 1);

    // --- GCN layer 1: prune top-1638 kappa nodes ---
    topk_mask<<<N / 256, 256, 0, stream>>>(KAPPA, MASK, N, NUM_PRUNE);
    degree_kernel<<<N / 256, 256, 0, stream>>>(adj, MASK, DINV, N);
    gemm_wmma_bf16<<<gemmGridCls, gemmBlk, 0, stream>>>(HBUF, W1, nullptr, ZBUF,
                                                        N, NHID, NCLASS, 0);
    aggregate<<<N, 64, 0, stream>>>(adj, MASK, DINV, ZBUF, b1, H2, N, NCLASS, 1);

    // --- log_softmax over classes ---
    log_softmax_k<<<N, 64, 0, stream>>>(H2, out, NCLASS);
}
